// ScaledDotProductAttention_41274635714572
// MI455X (gfx1250) — matrix operand order repaired and checked by reading, not compiled, not run on hardware
//
#include <hip/hip_runtime.h>

typedef __attribute__((ext_vector_type(16))) __bf16 v16bf;
typedef __attribute__((ext_vector_type(8)))  float  v8f;
typedef __attribute__((ext_vector_type(4)))  float  fvec4;

#define B_   4
#define H_   16
#define S_   2048
#define D_   64
#define SROW 2052              // padded score row stride (floats): 4-bank skew per row
#define VSTR 40                // vT row stride in bf16 (80B, 16B-aligned, 20-bank skew)
#define LDS_BYTES (16*SROW*4 + 2*64*VSTR*2 + 128*4 + 16*4)

__global__ __launch_bounds__(128) void sdpa_gfx1250_kernel(
    const float* __restrict__ q, const float* __restrict__ k,
    const float* __restrict__ v, const int* __restrict__ mask,
    float* __restrict__ out, float* __restrict__ attn)
{
    extern __shared__ __align__(16) char smem[];
    float*  sS   = (float*)smem;                               // [16][SROW] fp32 scores/probs
    __bf16* vT   = (__bf16*)(smem + 16*SROW*4);                // [2][64][VSTR] bf16 V^T tiles
    float*  red  = (float*)(smem + 16*SROW*4 + 2*64*VSTR*2);   // [8*16] partial reductions
    float*  rowv = red + 128;                                  // [16] row max -> row inv-sum

    const int tid   = threadIdx.x;
    const int lane  = tid & 31;
    const int w     = tid >> 5;         // wave id 0..3
    const int n     = lane & 15;        // matrix row (A) / col (B,C)
    const int hi    = lane >> 4;        // lane-half select
    const int base8 = hi << 3;          // C/D row offset (M = r + 8*hi)
    const int hi16  = hi << 4;          // B-fragment K split

    const int wg    = blockIdx.x;
    const int qblk  = wg & (S_/16 - 1); // 128 query blocks per (b,h)
    const int bh    = wg >> 7;
    const int b     = bh >> 4;
    const int qbase = qblk << 4;

    const float* qp = q + (size_t)bh * S_ * D_;
    const float* kp = k + (size_t)bh * S_ * D_;
    const float* vp = v + (size_t)bh * S_ * D_;
    const int*   mp = mask + (size_t)b * S_ * S_;

    // ---------------- Phase 1: S = (Q * 1/sqrt(64)) @ K^T, masked ----------------
    v16bf aq0, aq1;
    {
        const float* qrow = qp + (size_t)(qbase + n) * D_;
        #pragma unroll
        for (int j = 0; j < 8; ++j) {
            aq0[j]     = (__bf16)(qrow[hi16 + j]           * 0.125f);
            aq0[j + 8] = (__bf16)(qrow[hi16 + 8 + j]      * 0.125f);
            aq1[j]     = (__bf16)(qrow[32 + hi16 + j]      * 0.125f);
            aq1[j + 8] = (__bf16)(qrow[32 + hi16 + 8 + j] * 0.125f);
        }
    }
    for (int kt = w; kt < S_/16; kt += 4) {
        const int kb = kt << 4;
        const float* krow = kp + (size_t)(kb + n) * D_;   // B col n = K row kb+n
        v16bf bk0, bk1;
        #pragma unroll
        for (int j = 0; j < 16; ++j) {
            bk0[j] = (__bf16)krow[hi16 + j];
            bk1[j] = (__bf16)krow[32 + hi16 + j];
        }
        v8f c = {};
        c = __builtin_amdgcn_wmma_f32_16x16x32_bf16(false, aq0, false, bk0, (short)0, c, false, false);
        c = __builtin_amdgcn_wmma_f32_16x16x32_bf16(false, aq1, false, bk1, (short)0, c, false, false);
        #pragma unroll
        for (int r = 0; r < 8; ++r) {
            const int row  = r + base8;                    // C layout: M = r + 8*hi
            const int mval = mp[(size_t)(qbase + row) * S_ + kb + n];
            sS[row * SROW + kb + n] = (mval == 0) ? -1.0e9f : c[r];
        }
    }
    __syncthreads();

    // ---------------- Phase 2: row softmax (fp32) ----------------
    {
        const int row = tid & 15;
        const int sl  = tid >> 4;                          // 8 slices of 256 cols
        float* srow = sS + row * SROW + sl * 256;
        float mx = -3.0e38f;
        #pragma unroll 8
        for (int cidx = 0; cidx < 256; ++cidx) mx = fmaxf(mx, srow[cidx]);
        red[sl * 16 + row] = mx;
        __syncthreads();
        if (tid < 16) {
            float m2 = red[tid];
            #pragma unroll
            for (int s = 1; s < 8; ++s) m2 = fmaxf(m2, red[s * 16 + tid]);
            rowv[tid] = m2;
        }
        __syncthreads();
        const float rm = rowv[row];
        float sum = 0.0f;
        #pragma unroll 8
        for (int cidx = 0; cidx < 256; ++cidx) {
            float e = __expf(srow[cidx] - rm);
            srow[cidx] = e;
            sum += e;
        }
        red[sl * 16 + row] = sum;
        __syncthreads();
        if (tid < 16) {
            float s2 = 0.0f;
            #pragma unroll
            for (int s = 0; s < 8; ++s) s2 += red[s * 16 + tid];
            rowv[tid] = 1.0f / s2;
        }
        __syncthreads();
    }
    // Normalize in place + stream attn to HBM (non-temporal, fully coalesced b128).
    {
        float* attnp = attn + ((size_t)bh * S_ + qbase) * S_;
        #pragma unroll 4
        for (int i = 0; i < 64; ++i) {
            const int pos4 = i * 128 + tid;
            const int flat = pos4 << 2;
            const int row  = flat >> 11;
            const int col  = flat & (S_ - 1);
            const float inv = rowv[row];
            float* sp = sS + row * SROW + col;
            fvec4 pv = *(fvec4*)sp;
            pv *= inv;
            *(fvec4*)sp = pv;
            __builtin_nontemporal_store(pv, (fvec4*)(attnp + (size_t)row * S_ + col));
        }
    }
    __syncthreads();

    // ---------------- Phase 3: out = P @ V (software-pipelined V staging) ----------------
    const int dbase = w << 4;                              // wave owns 16 d-columns
    const int kr    = tid & 31;
    const int dq    = tid >> 5;

    // helpers (inlined lambdas)
    auto vload = [&](int kc, fvec4* f) {                   // global V[kc*32+kr][*] -> regs
        const float* vrow = vp + (size_t)((kc << 5) + kr) * D_;
        #pragma unroll
        for (int i = 0; i < 4; ++i) f[i] = *(const fvec4*)(vrow + ((dq + i * 4) << 2));
    };
    auto vstore = [&](int kc, const fvec4* f) {            // regs -> bf16 transposed vT tile
        __bf16* vt = vT + (kc & 1) * (64 * VSTR);
        #pragma unroll
        for (int i = 0; i < 4; ++i) {
            const int d0 = (dq + i * 4) << 2;
            vt[(d0 + 0) * VSTR + kr] = (__bf16)f[i][0];
            vt[(d0 + 1) * VSTR + kr] = (__bf16)f[i][1];
            vt[(d0 + 2) * VSTR + kr] = (__bf16)f[i][2];
            vt[(d0 + 3) * VSTR + kr] = (__bf16)f[i][3];
        }
    };
    auto pcompute = [&](int kc, v8f& acc_) {               // one 16x16x32 WMMA from LDS
        const int kb = kc << 5;
        const __bf16* vt = vT + (kc & 1) * (64 * VSTR);
        v16bf pa, vb;
        const float* prow = sS + n * SROW + kb + hi16;    // A: P row n, K contig
        #pragma unroll
        for (int j = 0; j < 8; ++j) {
            pa[j]     = (__bf16)prow[j];
            pa[j + 8] = (__bf16)prow[8 + j];
        }
        const __bf16* vtr = vt + (size_t)(dbase + n) * VSTR + hi16;  // B: col = d, K contig
        #pragma unroll
        for (int j = 0; j < 16; ++j) vb[j] = vtr[j];
        acc_ = __builtin_amdgcn_wmma_f32_16x16x32_bf16(false, pa, false, vb, (short)0, acc_, false, false);
    };

    v8f acc = {};
    {
        fvec4 f0[4];
        vload(0, f0);
        vstore(0, f0);
    }
    __syncthreads();
    for (int kc = 0; kc < S_/32 - 1; ++kc) {
        fvec4 nf[4];
        vload(kc + 1, nf);          // global prefetch overlaps WMMA below
        pcompute(kc, acc);          // consume buffer kc&1
        vstore(kc + 1, nf);         // fill buffer (kc+1)&1 (last read before prev barrier)
        __syncthreads();
    }
    pcompute(S_/32 - 1, acc);

    float* op = out + ((size_t)bh * S_ + qbase) * D_ + dbase + n;
    #pragma unroll
    for (int r = 0; r < 8; ++r)
        op[(size_t)(r + base8) * D_] = acc[r];
}

extern "C" void kernel_launch(void* const* d_in, const int* in_sizes, int n_in,
                              void* d_out, int out_size, void* d_ws, size_t ws_size,
                              hipStream_t stream) {
    (void)in_sizes; (void)n_in; (void)out_size; (void)d_ws; (void)ws_size;
    const float* q    = (const float*)d_in[0];
    const float* k    = (const float*)d_in[1];
    const float* v    = (const float*)d_in[2];
    const int*   mask = (const int*)d_in[3];
    float* out  = (float*)d_out;
    float* attn = out + (size_t)B_ * H_ * S_ * D_;   // tuple order: (out, attn)

    const int grid = B_ * H_ * (S_ / 16);            // 8192 workgroups
    sdpa_gfx1250_kernel<<<grid, 128, LDS_BYTES, stream>>>(q, k, v, mask, out, attn);
}
